// Model_16398185136955
// MI455X (gfx1250) — matrix operand order repaired and checked by reading, not compiled, not run on hardware
//
#include <hip/hip_runtime.h>
#include <hip/hip_bf16.h>

// ---------------------------------------------------------------------------
// Types for WMMA fragments (gfx1250, wave32)
// ---------------------------------------------------------------------------
typedef _Float16 half8 __attribute__((ext_vector_type(8)));
typedef _Float16 v16h  __attribute__((ext_vector_type(16)));
typedef float    v8f   __attribute__((ext_vector_type(8)));

union ABu { v16h v; half8 h[2]; };

// Problem constants
#define NSENT 4000
#define LSEQ  128
#define EMBP  64      // EMB=60 padded to 64 (one conv tap per 64-block of K)
#define KP    192     // 3 * 64
#define HID   230
#define HIDP  240     // padded to 15 * 16
#define NBAG  500
#define BAGSZ 8
#define NREL  25

#define A_BYTES   (HIDP * KP * 2)          // 92160  (Wt[hid][k] f16)
#define EMB_BYTES (130 * EMBP * 2)         // 16640  (rows 0 and 129 are zero pad)
#define HPART_BYTES (16 * HIDP * 4)        // 15360  (8 waves x 2 half-tiles)

// ---------------------------------------------------------------------------
// Kernel 0: repack conv_w [230,60,3] f32 -> Wt f16 [240][192], Wt[h][dl*64+e]
// ---------------------------------------------------------------------------
__global__ void prep_A_kernel(const float* __restrict__ conv_w, _Float16* __restrict__ A) {
    int idx = blockIdx.x * 256 + threadIdx.x;
    if (idx >= HIDP * KP) return;
    int row = idx / KP;
    int k   = idx % KP;
    int dl  = k >> 6;          // which conv tap (0..2)
    int e   = k & 63;          // embedding channel
    float v = (row < HID && e < 60) ? conv_w[(row * 60 + e) * 3 + dl] : 0.0f;
    A[idx] = (_Float16)v;
}

// ---------------------------------------------------------------------------
// Kernel 1: per-sentence gather + im2col-GEMM conv via WMMA + maxpool + ReLU
//   GEMM is D[l, hid] = emb_col[l, k] * Wt[k, hid]:  M = L, N = HIDP, K = 192.
//   grid = 4000 blocks, 256 threads (8 waves); wave w owns L(M)-tile w.
// ---------------------------------------------------------------------------
__global__ __launch_bounds__(256)
void conv_encoder_kernel(const int* __restrict__ X,
                         const int* __restrict__ pos1,
                         const int* __restrict__ pos2,
                         const float* __restrict__ word_emb,
                         const float* __restrict__ pos1_emb,
                         const float* __restrict__ pos2_emb,
                         const _Float16* __restrict__ Wglob,
                         const float* __restrict__ conv_b,
                         float* __restrict__ hOut) {
    extern __shared__ char smem[];
    _Float16* Wlds   = (_Float16*)smem;                              // [240][192]
    _Float16* embLds = (_Float16*)(smem + A_BYTES);                  // [130][64]
    float*    hpart  = (float*)(smem + A_BYTES + EMB_BYTES);         // [16][240]

    const int n   = blockIdx.x;
    const int tid = threadIdx.x;

    // --- stage Wt (broadcast weights) into LDS with 16B copies
    {
        const uint4* src = (const uint4*)Wglob;
        uint4* dst = (uint4*)Wlds;
        for (int i = tid; i < A_BYTES / 16; i += 256) dst[i] = src[i];
    }
    // --- zero the halo rows of the embedding tile (rows 0 and 129)
    if (tid < 64) {
        embLds[tid] = (_Float16)0.0f;
        embLds[129 * EMBP + tid] = (_Float16)0.0f;
    }
    // --- gather word/pos embeddings -> f16 LDS tile, row l+1, channels padded to 64
    for (int i = tid; i < LSEQ * EMBP; i += 256) {
        int l = i >> 6;
        int e = i & 63;
        float v = 0.0f;
        if (e < 50)      v = word_emb[X[n * LSEQ + l] * 50 + e];
        else if (e < 55) v = pos1_emb[pos1[n * LSEQ + l] * 5 + (e - 50)];
        else if (e < 60) v = pos2_emb[pos2[n * LSEQ + l] * 5 + (e - 55)];
        embLds[(l + 1) * EMBP + e] = (_Float16)v;
    }
    __syncthreads();

    const int wave = tid >> 5;
    const int lane = tid & 31;
    const int wt   = wave;            // this wave's 16-row L (M) tile
    const int lo16 = lane & 15;
    const int hi   = lane >> 4;

    // --- preload the 6 A fragments (im2col of emb) for this L-tile; reused for
    //     all 15 N-tiles.  A 16x32 f16 layout: lane<16: row M=lane, K {0..7,16..23};
    //     lane>=16: row M=lane-16, K {8..15,24..31}.  A[l][dl*64+e] = embLds[l+dl][e].
    v16h afrag[6];
    #pragma unroll
    for (int kt = 0; kt < 6; ++kt) {
        const _Float16* p = embLds
            + (wt * 16 + lo16 + (kt >> 1)) * EMBP   // row l + tap(dl)
            + (kt & 1) * 32 + hi * 8;               // K offset inside the 64-block
        ABu u;
        u.h[0] = *(const half8*)(p);
        u.h[1] = *(const half8*)(p + 16);
        afrag[kt] = u.v;
    }

    // --- GEMM: 15 N(hid)-tiles x 6 K-steps of v_wmma_f32_16x16x32_f16
    #pragma unroll 1
    for (int nt = 0; nt < 15; ++nt) {
        // B 32x16 f16 layout: lanes 0-15 hold K {0..7,16..23} (col N=lane),
        // lanes 16-31 hold K {8..15,24..31}.  B[k][hid] = Wt[hid][k] (contiguous in k).
        v16h bfr[6];
        #pragma unroll
        for (int kt = 0; kt < 6; ++kt) {
            const _Float16* p = Wlds + (nt * 16 + lo16) * KP + kt * 32 + hi * 8;
            ABu u;
            u.h[0] = *(const half8*)(p);
            u.h[1] = *(const half8*)(p + 16);
            bfr[kt] = u.v;
        }
        v8f acc = {};
        #pragma unroll
        for (int kt = 0; kt < 6; ++kt)
            acc = __builtin_amdgcn_wmma_f32_16x16x32_f16(
                false, afrag[kt], false, bfr[kt], (short)0, acc, false, false);

        // --- max over L entirely in-register: lane holds (hid = nt*16+lo16,
        //     l = wt*16 + r + 8*hi) for r = 0..7.  7 v_max ops, no cross-lane.
        float v = acc[0];
        #pragma unroll
        for (int r = 1; r < 8; ++r) v = fmaxf(v, acc[r]);
        hpart[(wave * 2 + hi) * HIDP + nt * 16 + lo16] = v;   // 32 distinct addrs
    }
    __syncthreads();

    // --- fold the 16 partial maxima, add bias, ReLU, store h[n]
    for (int t = tid; t < HID; t += 256) {
        float m = hpart[t];
        #pragma unroll
        for (int w = 1; w < 16; ++w) m = fmaxf(m, hpart[w * HIDP + t]);
        hOut[n * HID + t] = fmaxf(m + conv_b[t], 0.0f);
    }
}

// ---------------------------------------------------------------------------
// Kernel 2: per-bag softmax attention pooling + relation classifier
//   grid = 500 blocks, 256 threads (8 waves, one wave per bag member).
// ---------------------------------------------------------------------------
__global__ __launch_bounds__(256)
void bag_attn_kernel(const float* __restrict__ h,
                     const int* __restrict__ relation,
                     const float* __restrict__ rel_w,
                     const float* __restrict__ rel_b,
                     float* __restrict__ out) {
    __shared__ float hs[BAGSZ][232];
    __shared__ float q[232];
    __shared__ float logits[BAGSZ];
    __shared__ float alpha[BAGSZ];
    __shared__ float brep[232];

    const int bag = blockIdx.x;
    const int tid = threadIdx.x;
    const int rel = relation[bag];

    for (int i = tid; i < BAGSZ * HID; i += 256)
        hs[i / HID][i % HID] = h[bag * BAGSZ * HID + i];
    for (int i = tid; i < HID; i += 256)
        q[i] = rel_w[rel * HID + i];
    __syncthreads();

    const int wave = tid >> 5;
    const int lane = tid & 31;

    // attention logit for sentence `wave` of this bag
    {
        float s = 0.0f;
        for (int k = lane; k < HID; k += 32) s += hs[wave][k] * q[k];
        for (int off = 16; off; off >>= 1) s += __shfl_down(s, off);
        if (lane == 0) logits[wave] = s;
    }
    __syncthreads();

    if (tid == 0) {
        float m = logits[0];
        #pragma unroll
        for (int i = 1; i < BAGSZ; ++i) m = fmaxf(m, logits[i]);
        float e[BAGSZ];
        float d = 0.0f;
        #pragma unroll
        for (int i = 0; i < BAGSZ; ++i) { e[i] = __expf(logits[i] - m); d += e[i]; }
        float inv = 1.0f / d;
        #pragma unroll
        for (int i = 0; i < BAGSZ; ++i) alpha[i] = e[i] * inv;
    }
    __syncthreads();

    // weighted bag representation
    for (int i = tid; i < HID; i += 256) {
        float s = 0.0f;
        #pragma unroll
        for (int j = 0; j < BAGSZ; ++j) s += alpha[j] * hs[j][i];
        brep[i] = s;
    }
    __syncthreads();

    // classifier: out[bag, r] = brep . rel_w[r] + rel_b[r]
    for (int r = wave; r < NREL; r += 8) {
        float s = 0.0f;
        for (int k = lane; k < HID; k += 32) s += brep[k] * rel_w[r * HID + k];
        for (int off = 16; off; off >>= 1) s += __shfl_down(s, off);
        if (lane == 0) out[bag * NREL + r] = s + rel_b[r];
    }
}

// ---------------------------------------------------------------------------
// Launch
// ---------------------------------------------------------------------------
extern "C" void kernel_launch(void* const* d_in, const int* in_sizes, int n_in,
                              void* d_out, int out_size, void* d_ws, size_t ws_size,
                              hipStream_t stream) {
    (void)in_sizes; (void)n_in; (void)out_size; (void)ws_size;
    const int*   X        = (const int*)d_in[0];
    const int*   pos1     = (const int*)d_in[1];
    const int*   pos2     = (const int*)d_in[2];
    // d_in[3] mask, d_in[4] length, d_in[5] scope: unused (bags are contiguous, size 8)
    const int*   relation = (const int*)d_in[6];
    const float* word_emb = (const float*)d_in[7];
    const float* pos1_emb = (const float*)d_in[8];
    const float* pos2_emb = (const float*)d_in[9];
    const float* conv_w   = (const float*)d_in[10];
    const float* conv_b   = (const float*)d_in[11];
    const float* rel_w    = (const float*)d_in[12];
    const float* rel_b    = (const float*)d_in[13];

    _Float16* Wf16 = (_Float16*)d_ws;                       // 92160 B
    float*    hBuf = (float*)((char*)d_ws + A_BYTES);       // 4000*230*4 B
    float*    out  = (float*)d_out;                         // [500, 25]

    prep_A_kernel<<<(HIDP * KP + 255) / 256, 256, 0, stream>>>(conv_w, Wf16);

    size_t lds = A_BYTES + EMB_BYTES + HPART_BYTES;         // 124160 B
    conv_encoder_kernel<<<NSENT, 256, lds, stream>>>(
        X, pos1, pos2, word_emb, pos1_emb, pos2_emb, Wf16, conv_b, hBuf);

    bag_attn_kernel<<<NBAG, 256, 0, stream>>>(hBuf, relation, rel_w, rel_b, out);
}
